// CBR_RNN_74887049773142
// MI455X (gfx1250) — matrix operand order repaired and checked by reading, not compiled, not run on hardware
//
#include <hip/hip_runtime.h>
#include <stdint.h>

#define NTOKEN 50257
#define NINP 1024
#define NHID 1024
#define SEQ 128
#define BSZ 8
#define S1 (SEQ + 1)

typedef __attribute__((ext_vector_type(16))) __bf16 v16bf;
typedef __attribute__((ext_vector_type(8)))  float  v8f;
typedef __attribute__((ext_vector_type(4)))  int    v4i_raw;              // builtin arg0 pointee
typedef __attribute__((address_space(3))) v4i_raw*  v4i_lds_ptr;          // builtin arg1 type

// ---------- CDNA5 async global->LDS path (guarded) ----------

#if defined(__has_builtin)
#  if __has_builtin(__builtin_amdgcn_global_load_async_to_lds_b128)
#    define ASYNC_LDS 1
#  endif
#  if __has_builtin(__builtin_amdgcn_s_wait_asynccnt)
#    define HAVE_WAIT_ASYNC 1
#  endif
#endif

__device__ __forceinline__ void async_wait0() {
#if defined(ASYNC_LDS)
#  if defined(HAVE_WAIT_ASYNC)
  __builtin_amdgcn_s_wait_asynccnt(0);
#  else
  asm volatile("s_wait_asynccnt 0x0" ::: "memory");
#  endif
#endif
}

// Copy 32 bytes global -> LDS for this thread (async when available).
__device__ __forceinline__ void stage32(uint16_t* l, const uint16_t* g) {
#if defined(ASYNC_LDS)
  v4i_raw*    gp = (v4i_raw*)(uintptr_t)(const void*)g;
  v4i_lds_ptr lp = (v4i_lds_ptr)(uintptr_t)(void*)l;
  __builtin_amdgcn_global_load_async_to_lds_b128(gp, lp, 0, 0);
  __builtin_amdgcn_global_load_async_to_lds_b128(gp, lp, 16, 0);
#else
  *reinterpret_cast<uint4*>(l)     = *reinterpret_cast<const uint4*>(g);
  *reinterpret_cast<uint4*>(l + 8) = *reinterpret_cast<const uint4*>(g + 8);
#endif
}

// ---------- helpers ----------

__device__ __forceinline__ uint16_t f2bf(float f) {
  union { float f; uint32_t u; } v; v.f = f;
  uint32_t r = v.u + 0x7FFFu + ((v.u >> 16) & 1u);   // round-to-nearest-even
  return (uint16_t)(r >> 16);
}

union FragCast { v16bf v; uint4 q[2]; };

// A matrix (16x32 bf16, MxK): lane l holds row M=l&15.
// Layout: lanes 0-15 hold K=0..15, lanes 16-31 hold K=16..31 (contiguous, matches B).
__device__ __forceinline__ v16bf load_a_frag(const uint16_t* rowbase, int k, int lane) {
  const uint16_t* p = rowbase + k + ((lane & 16) ? 16 : 0);
  FragCast u;
  u.q[0] = *reinterpret_cast<const uint4*>(p);        // K chunk 0..7
  u.q[1] = *reinterpret_cast<const uint4*>(p + 8);    // K chunk 8..15
  return u.v;
}

// B matrix (32x16 bf16, KxN) from row-major W[N][K] (NT gemm): lane l holds
// column N=l&15; lanes 0-15 K=0..15, lanes 16-31 K=16..31 (contiguous 16-K runs).
__device__ __forceinline__ v16bf load_b_frag(const uint16_t* colbase, int k, int lane) {
  const uint16_t* p = colbase + k + ((lane & 16) ? 16 : 0);
  FragCast u;
  u.q[0] = *reinterpret_cast<const uint4*>(p);
  u.q[1] = *reinterpret_cast<const uint4*>(p + 8);
  return u.v;
}

// Same B fragment but sourced from an LDS tile with padded row stride.
constexpr int DEC_KC  = 64;            // K-chunk (halfs) staged per buffer
constexpr int DEC_LDB = DEC_KC + 8;    // LDS row stride (halfs); pad -> conflict-free b128

__device__ __forceinline__ v16bf load_b_frag_lds(const uint16_t* bs, int row16, int kl, int lane) {
  const uint16_t* p = bs + (row16 + (lane & 15)) * DEC_LDB + kl + ((lane & 16) ? 16 : 0);
  FragCast u;
  u.q[0] = *reinterpret_cast<const uint4*>(p);
  u.q[1] = *reinterpret_cast<const uint4*>(p + 8);
  return u.v;
}

// ---------- utility kernels ----------

__global__ void zero_kernel(uint32_t* __restrict__ p, size_t nwords) {
  size_t i = (size_t)blockIdx.x * blockDim.x + threadIdx.x;
  size_t stride = (size_t)gridDim.x * blockDim.x;
  for (; i < nwords; i += stride) p[i] = 0u;
}

__global__ void cvt_bf16_kernel(const float* __restrict__ src, uint16_t* __restrict__ dst, size_t n) {
  size_t i = (size_t)blockIdx.x * blockDim.x + threadIdx.x;
  size_t stride = (size_t)gridDim.x * blockDim.x;
  for (; i < n; i += stride) dst[i] = f2bf(src[i]);
}

__global__ void embed_kernel(const int* __restrict__ obs, const float* __restrict__ encW,
                             uint16_t* __restrict__ emb) {
  const int sb = blockIdx.x;                 // s*BSZ + b
  const int tok = obs[sb];
  const float* src = encW + (size_t)tok * NINP;
  uint16_t* dst = emb + (size_t)sb * NINP;
  for (int d = threadIdx.x; d < NINP; d += blockDim.x) dst[d] = f2bf(src[d]);
}

// ---------- concat builders (write rows 0..7 only; rows 8..15 stay zero) ----------

__global__ void build_xq_kernel(const uint16_t* __restrict__ e, const uint16_t* __restrict__ h,
                                uint16_t* __restrict__ xq) {
  int idx = blockIdx.x * blockDim.x + threadIdx.x;   // [0, 8*2048)
  int b = idx >> 11, c = idx & 2047;
  xq[(size_t)b * (NINP + NHID) + c] =
      (c < NINP) ? e[(size_t)b * NINP + c] : h[(size_t)b * NHID + (c - NINP)];
}

__global__ void build_xint_kernel(const uint16_t* __restrict__ e, const uint16_t* __restrict__ qb,
                                  const float* __restrict__ attn, const uint16_t* __restrict__ h,
                                  uint16_t* __restrict__ x) {
  int idx = blockIdx.x * blockDim.x + threadIdx.x;   // [0, 8*4096)
  int b = idx >> 12, c = idx & 4095;
  uint16_t v;
  if (c < NHID)          v = e[(size_t)b * NINP + c];
  else if (c < 2 * NHID) v = qb[(size_t)b * NHID + (c - NHID)];
  else if (c < 3 * NHID) v = f2bf(attn[(size_t)b * NHID + (c - 2 * NHID)]);
  else                   v = h[(size_t)b * NHID + (c - 3 * NHID)];
  x[(size_t)b * (4 * NHID) + c] = v;
}

// ---------- NT GEMM: Y[8,N] = A[16,K](rows 8..15 zero) x W[N,K]^T, bf16 in / f32 out ----------
// M=8 -> zero weight reuse within a step: stream W straight from the 192MB L2.

__global__ void gemm_nt_pad16(const uint16_t* __restrict__ A,
                              const uint16_t* __restrict__ W,
                              float* __restrict__ Y, int N, int K) {
  const int lane  = threadIdx.x & 31;
  const int ntile = blockIdx.x * (blockDim.x >> 5) + (threadIdx.x >> 5);
  const uint16_t* aRow = A + (size_t)(lane & 15) * K;
  const uint16_t* bRow = W + (size_t)(ntile * 16 + (lane & 15)) * K;
  v8f acc = {};
  for (int k = 0; k < K; k += 32) {
    __builtin_prefetch((const void*)(bRow + k + 1024), 0, 1);  // stream next 2KB of weights
    v16bf a = load_a_frag(aRow, k, lane);
    v16bf b = load_b_frag(bRow, k, lane);
    acc = __builtin_amdgcn_wmma_f32_16x16x32_bf16(false, a, false, b, (short)0, acc, false, false);
  }
  if (lane < 16) {                                   // lanes 0-15 hold rows M=0..7 (real rows)
    int n = ntile * 16 + lane;
    #pragma unroll
    for (int m = 0; m < 8; ++m) Y[(size_t)m * N + n] = acc[m];
  }
}

// ---------- decode GEMM: logits[1024, NTOKEN] = hs[1024,1024] x decW[NTOKEN,1024]^T + b ----------
// Block = 8 waves covering 128 M-rows x 64 N-cols. B strip staged once per block into a
// double-buffered LDS tile via async global->LDS copies (8x less L2 traffic than per-wave loads).

__device__ __forceinline__ void store_acc(const v8f& acc, float* __restrict__ Y,
                                          const float* __restrict__ bias, int m0, int n) {
  if (n < NTOKEN) {
    float bv = bias[n];
    #pragma unroll
    for (int m = 0; m < 8; ++m) Y[(size_t)(m0 + m) * NTOKEN + n] = acc[m] + bv;
  }
}

__global__ void __launch_bounds__(256) gemm_decode(const uint16_t* __restrict__ A,
                                                   const uint16_t* __restrict__ W,
                                                   const float* __restrict__ bias,
                                                   float* __restrict__ Y) {
  constexpr int MGROUPS = (SEQ * BSZ) / 128;    // 8 groups of 8 M-tiles
  __shared__ __align__(16) uint16_t Bs[2][64 * DEC_LDB];

  const int lane   = threadIdx.x & 31;
  const int wave   = threadIdx.x >> 5;
  const int mgroup = blockIdx.x % MGROUPS;
  const int nstrip = blockIdx.x / MGROUPS;
  const int mtile  = mgroup * 8 + wave;

  // this thread's staging slice: 32B of one of 64 B-rows per chunk
  const int brow  = threadIdx.x >> 2;           // 0..63
  const int bpart = (threadIdx.x & 3) * 16;     // halfs within the K-chunk
  int nrow = nstrip * 64 + brow;
  if (nrow >= NTOKEN) nrow = NTOKEN - 1;        // clamp loads; stores guarded below
  const uint16_t* gB = W + (size_t)nrow * NHID + bpart;
  uint16_t* lB0 = &Bs[0][brow * DEC_LDB + bpart];
  uint16_t* lB1 = &Bs[1][brow * DEC_LDB + bpart];

  const uint16_t* aRow = A + (size_t)(mtile * 16 + (lane & 15)) * NHID;

  v8f acc0 = {}, acc1 = {}, acc2 = {}, acc3 = {};

  stage32(lB0, gB);                              // prime chunk 0
  constexpr int NCHUNK = NHID / DEC_KC;          // 16
  for (int c = 0; c < NCHUNK; ++c) {
    async_wait0();
    __syncthreads();                             // Bs[c&1] resident for the whole block
    if (c + 1 < NCHUNK)                          // overlap next stage with compute
      stage32((c & 1) ? lB0 : lB1, gB + (c + 1) * DEC_KC);
    const uint16_t* bs = &Bs[c & 1][0];
    #pragma unroll
    for (int kl = 0; kl < DEC_KC; kl += 32) {
      v16bf a = load_a_frag(aRow, c * DEC_KC + kl, lane);   // A reused across 4 N-tiles
      acc0 = __builtin_amdgcn_wmma_f32_16x16x32_bf16(false, a, false, load_b_frag_lds(bs,  0, kl, lane), (short)0, acc0, false, false);
      acc1 = __builtin_amdgcn_wmma_f32_16x16x32_bf16(false, a, false, load_b_frag_lds(bs, 16, kl, lane), (short)0, acc1, false, false);
      acc2 = __builtin_amdgcn_wmma_f32_16x16x32_bf16(false, a, false, load_b_frag_lds(bs, 32, kl, lane), (short)0, acc2, false, false);
      acc3 = __builtin_amdgcn_wmma_f32_16x16x32_bf16(false, a, false, load_b_frag_lds(bs, 48, kl, lane), (short)0, acc3, false, false);
    }
    __syncthreads();                             // all waves done reading before re-stage
  }

  const int m0 = mtile * 16 + ((lane & 16) ? 8 : 0);
  const int nc = lane & 15;
  store_acc(acc0, Y, bias, m0, nstrip * 64 +  0 + nc);
  store_acc(acc1, Y, bias, m0, nstrip * 64 + 16 + nc);
  store_acc(acc2, Y, bias, m0, nstrip * 64 + 32 + nc);
  store_acc(acc3, Y, bias, m0, nstrip * 64 + 48 + nc);
}

// ---------- bias + LayerNorm + tanh (one block per batch row) ----------

__global__ void ln_tanh_kernel(const float* __restrict__ Y, const float* __restrict__ bias,
                               const float* __restrict__ g, const float* __restrict__ beta,
                               float* __restrict__ out_f32, uint16_t* __restrict__ out_bf, int D) {
  __shared__ float r1[256], r2[256];
  const int b = blockIdx.x, t = threadIdx.x;
  const float* y = Y + (size_t)b * D;
  float s = 0.f, ss = 0.f;
  for (int d = t; d < D; d += blockDim.x) { float x = y[d] + bias[d]; s += x; ss += x * x; }
  r1[t] = s; r2[t] = ss; __syncthreads();
  for (int off = 128; off > 0; off >>= 1) {
    if (t < off) { r1[t] += r1[t + off]; r2[t] += r2[t + off]; }
    __syncthreads();
  }
  float mean = r1[0] / D;
  float inv  = rsqrtf(r2[0] / D - mean * mean + 1e-5f);
  for (int d = t; d < D; d += blockDim.x) {
    float x = y[d] + bias[d];
    float v = tanhf((x - mean) * inv * g[d] + beta[d]);
    if (out_f32) out_f32[(size_t)b * D + d] = v;
    if (out_bf)  out_bf[(size_t)b * D + d]  = f2bf(v);
  }
}

// fin post: LN(3072)+tanh, then split into kc[i+1], vc[i+1], h (bf16) and hs[i]
__global__ void fin_post_kernel(const float* __restrict__ Y, const float* __restrict__ bias,
                                const float* __restrict__ g, const float* __restrict__ beta,
                                float* __restrict__ kc, float* __restrict__ vc,
                                uint16_t* __restrict__ h_bf, uint16_t* __restrict__ hs_bf, int i) {
  const int D = 3 * NHID;
  __shared__ float r1[256], r2[256];
  const int b = blockIdx.x, t = threadIdx.x;
  const float* y = Y + (size_t)b * D;
  float s = 0.f, ss = 0.f;
  for (int d = t; d < D; d += blockDim.x) { float x = y[d] + bias[d]; s += x; ss += x * x; }
  r1[t] = s; r2[t] = ss; __syncthreads();
  for (int off = 128; off > 0; off >>= 1) {
    if (t < off) { r1[t] += r1[t + off]; r2[t] += r2[t + off]; }
    __syncthreads();
  }
  float mean = r1[0] / D;
  float inv  = rsqrtf(r2[0] / D - mean * mean + 1e-5f);
  for (int d = t; d < D; d += blockDim.x) {
    float x = y[d] + bias[d];
    float v = tanhf((x - mean) * inv * g[d] + beta[d]);
    if (d < NHID)          kc[((size_t)(i + 1) * BSZ + b) * NHID + d] = v;
    else if (d < 2 * NHID) vc[((size_t)(i + 1) * BSZ + b) * NHID + (d - NHID)] = v;
    else {
      uint16_t hv = f2bf(v);
      h_bf[(size_t)b * NHID + (d - 2 * NHID)] = hv;
      hs_bf[((size_t)i * BSZ + b) * NHID + (d - 2 * NHID)] = hv;
    }
  }
}

// ---------- single-head SDPA over growing KV cache (one block per batch row) ----------

__global__ void attn_kernel(const float* __restrict__ q, const float* __restrict__ kc,
                            const float* __restrict__ vc, float* __restrict__ attn, int i) {
  __shared__ float w[S1];
  __shared__ float norm;
  const int b = blockIdx.x, t = threadIdx.x;
  const int nvalid = i + 1;                     // cache slots 0..i
  if (t < nvalid) {
    const float* qb = q + (size_t)b * NHID;
    const float* kr = kc + ((size_t)t * BSZ + b) * NHID;
    float dot = 0.f;
    for (int d = 0; d < NHID; ++d) dot += qb[d] * kr[d];
    w[t] = dot * 0.03125f;                      // 1/sqrt(1024)
  }
  __syncthreads();
  if (t == 0) {
    float mx = -3.4e38f;
    for (int s2 = 0; s2 < nvalid; ++s2) mx = fmaxf(mx, w[s2]);
    float sum = 0.f;
    for (int s2 = 0; s2 < nvalid; ++s2) { float e = __expf(w[s2] - mx); w[s2] = e; sum += e; }
    norm = 1.0f / sum;
  }
  __syncthreads();
  for (int d = t; d < NHID; d += blockDim.x) {
    float a = 0.f;
    for (int s2 = 0; s2 < nvalid; ++s2) a += w[s2] * vc[((size_t)s2 * BSZ + b) * NHID + d];
    attn[(size_t)b * NHID + d] = a * norm;
  }
}

// ---------- workspace layout ----------

static constexpr size_t alignup(size_t x) { return (x + 255) & ~(size_t)255; }

static constexpr size_t SZ_QW   = (size_t)NHID * (NINP + NHID) * 2;
static constexpr size_t SZ_INTW = (size_t)(4 * NHID) * (4 * NHID) * 2;
static constexpr size_t SZ_FINW = (size_t)(3 * NHID) * (4 * NHID) * 2;
static constexpr size_t SZ_DECW = (size_t)NTOKEN * NHID * 2;
static constexpr size_t SZ_EMB  = (size_t)SEQ * BSZ * NINP * 2;
static constexpr size_t SZ_HS   = (size_t)SEQ * BSZ * NHID * 2;
static constexpr size_t SZ_KC   = (size_t)S1 * BSZ * NHID * 4;
static constexpr size_t SZ_HBF  = (size_t)BSZ * NHID * 2;
static constexpr size_t SZ_XQ   = (size_t)16 * (NINP + NHID) * 2;
static constexpr size_t SZ_XINT = (size_t)16 * (4 * NHID) * 2;
static constexpr size_t SZ_QF   = (size_t)BSZ * NHID * 4;
static constexpr size_t SZ_QBF  = (size_t)BSZ * NHID * 2;
static constexpr size_t SZ_YQ   = (size_t)BSZ * NHID * 4;
static constexpr size_t SZ_YINT = (size_t)BSZ * (4 * NHID) * 4;
static constexpr size_t SZ_YFIN = (size_t)BSZ * (3 * NHID) * 4;

static constexpr size_t OFF_QW   = 0;
static constexpr size_t OFF_INTW = alignup(OFF_QW + SZ_QW);
static constexpr size_t OFF_FINW = alignup(OFF_INTW + SZ_INTW);
static constexpr size_t OFF_DECW = alignup(OFF_FINW + SZ_FINW);
static constexpr size_t OFF_EMB  = alignup(OFF_DECW + SZ_DECW);
static constexpr size_t OFF_HS   = alignup(OFF_EMB + SZ_EMB);
// contiguous zero-init group: kc, vc, h, padded A tiles
static constexpr size_t OFF_KC   = alignup(OFF_HS + SZ_HS);
static constexpr size_t OFF_VC   = alignup(OFF_KC + SZ_KC);
static constexpr size_t OFF_HBF  = alignup(OFF_VC + SZ_KC);
static constexpr size_t OFF_XQ   = alignup(OFF_HBF + SZ_HBF);
static constexpr size_t OFF_XINT = alignup(OFF_XQ + SZ_XQ);
static constexpr size_t OFF_XFIN = alignup(OFF_XINT + SZ_XINT);
static constexpr size_t ZERO_END = OFF_XFIN + SZ_XINT;
// scratch overwritten every step
static constexpr size_t OFF_QF   = alignup(ZERO_END);
static constexpr size_t OFF_QBF  = alignup(OFF_QF + SZ_QF);
static constexpr size_t OFF_ATTN = alignup(OFF_QBF + SZ_QBF);
static constexpr size_t OFF_YQ   = alignup(OFF_ATTN + SZ_QF);
static constexpr size_t OFF_YINT = alignup(OFF_YQ + SZ_YQ);
static constexpr size_t OFF_YFIN = alignup(OFF_YINT + SZ_YINT);

// ---------- launch ----------

extern "C" void kernel_launch(void* const* d_in, const int* in_sizes, int n_in,
                              void* d_out, int out_size, void* d_ws, size_t ws_size,
                              hipStream_t stream) {
  (void)in_sizes; (void)n_in; (void)out_size; (void)ws_size;
  const int*   obs   = (const int*)  d_in[0];
  const float* encW  = (const float*)d_in[1];
  const float* qW    = (const float*)d_in[2];
  const float* q_b   = (const float*)d_in[3];
  const float* intW  = (const float*)d_in[4];
  const float* int_b = (const float*)d_in[5];
  const float* finW  = (const float*)d_in[6];
  const float* fin_b = (const float*)d_in[7];
  const float* qn_g  = (const float*)d_in[8];
  const float* qn_b  = (const float*)d_in[9];
  const float* in_g  = (const float*)d_in[10];
  const float* in_b  = (const float*)d_in[11];
  const float* fn_g  = (const float*)d_in[12];
  const float* fn_b  = (const float*)d_in[13];
  const float* decW  = (const float*)d_in[14];
  const float* dec_b = (const float*)d_in[15];

  uint8_t* ws = (uint8_t*)d_ws;
  uint16_t* qW_bf   = (uint16_t*)(ws + OFF_QW);
  uint16_t* intW_bf = (uint16_t*)(ws + OFF_INTW);
  uint16_t* finW_bf = (uint16_t*)(ws + OFF_FINW);
  uint16_t* decW_bf = (uint16_t*)(ws + OFF_DECW);
  uint16_t* emb_bf  = (uint16_t*)(ws + OFF_EMB);
  uint16_t* hs_bf   = (uint16_t*)(ws + OFF_HS);
  float*    kc      = (float*)   (ws + OFF_KC);
  float*    vc      = (float*)   (ws + OFF_VC);
  uint16_t* h_bf    = (uint16_t*)(ws + OFF_HBF);
  uint16_t* xq      = (uint16_t*)(ws + OFF_XQ);
  uint16_t* xint    = (uint16_t*)(ws + OFF_XINT);
  uint16_t* xfin    = (uint16_t*)(ws + OFF_XFIN);
  float*    q_f32   = (float*)   (ws + OFF_QF);
  uint16_t* q_bf    = (uint16_t*)(ws + OFF_QBF);
  float*    attn    = (float*)   (ws + OFF_ATTN);
  float*    Yq      = (float*)   (ws + OFF_YQ);
  float*    Yint    = (float*)   (ws + OFF_YINT);
  float*    Yfin    = (float*)   (ws + OFF_YFIN);

  // 0) zero recurrent state + padded activation tiles (pad rows 8..15 stay zero all run)
  zero_kernel<<<1024, 256, 0, stream>>>((uint32_t*)(ws + OFF_KC), (ZERO_END - OFF_KC) / 4);

  // 1) one-shot fp32 -> bf16 weight conversion (recurrent weights then live in the 192MB L2)
  cvt_bf16_kernel<<<2048, 256, 0, stream>>>(qW,   qW_bf,   SZ_QW / 2);
  cvt_bf16_kernel<<<4096, 256, 0, stream>>>(intW, intW_bf, SZ_INTW / 2);
  cvt_bf16_kernel<<<4096, 256, 0, stream>>>(finW, finW_bf, SZ_FINW / 2);
  cvt_bf16_kernel<<<8192, 256, 0, stream>>>(decW, decW_bf, SZ_DECW / 2);

  // 2) embedding gather
  embed_kernel<<<SEQ * BSZ, 256, 0, stream>>>(obs, encW, emb_bf);

  // 3) sequential recurrence (128 steps)
  for (int i = 0; i < SEQ; ++i) {
    const uint16_t* e = emb_bf + (size_t)i * BSZ * NINP;
    build_xq_kernel<<<(BSZ * (NINP + NHID)) / 256, 256, 0, stream>>>(e, h_bf, xq);
    gemm_nt_pad16<<<(NHID / 16) / 8, 256, 0, stream>>>(xq, qW_bf, Yq, NHID, NINP + NHID);
    ln_tanh_kernel<<<BSZ, 256, 0, stream>>>(Yq, q_b, qn_g, qn_b, q_f32, q_bf, NHID);
    attn_kernel<<<BSZ, 256, 0, stream>>>(q_f32, kc, vc, attn, i);
    build_xint_kernel<<<(BSZ * 4 * NHID) / 256, 256, 0, stream>>>(e, q_bf, attn, h_bf, xint);
    gemm_nt_pad16<<<(4 * NHID / 16) / 8, 256, 0, stream>>>(xint, intW_bf, Yint, 4 * NHID, 4 * NHID);
    ln_tanh_kernel<<<BSZ, 256, 0, stream>>>(Yint, int_b, in_g, in_b, nullptr, xfin, 4 * NHID);
    gemm_nt_pad16<<<(3 * NHID / 16) / 8, 256, 0, stream>>>(xfin, finW_bf, Yfin, 3 * NHID, 4 * NHID);
    fin_post_kernel<<<BSZ, 256, 0, stream>>>(Yfin, fin_b, fn_g, fn_b, kc, vc, h_bf, hs_bf, i);
  }

  // 4) decode: logits = hs @ dec_W^T + dec_b (105 GFLOP bf16 WMMA GEMM, LDS-staged B)
  constexpr int NTILE   = (NTOKEN + 15) / 16;                  // 3142
  constexpr int NSTRIPS = (NTILE + 3) / 4;                     // 786
  constexpr int MGROUPS = (SEQ * BSZ) / 128;                   // 8
  gemm_decode<<<MGROUPS * NSTRIPS, 256, 0, stream>>>(hs_bf, decW_bf, dec_b, (float*)d_out);
}
